// Net_34428457845336
// MI455X (gfx1250) — matrix operand order repaired and checked by reading, not compiled, not run on hardware
//
#include <hip/hip_runtime.h>
#include <hip/hip_bf16.h>
#include <stdint.h>

typedef __attribute__((ext_vector_type(16))) _Float16 v16h;
typedef __attribute__((ext_vector_type(8)))  _Float16 v8h;
typedef __attribute__((ext_vector_type(8)))  float    v8f;

#define N_NODES 50000
#define N_EDGES 600000
#define NHID    128
#define NGRAPHS 512
#define NCLS    10
#define BN_EPS  1e-5f

// ---------------- utility kernels ----------------
__global__ void fill_f32(float* __restrict__ p, float v, int n) {
  int i = blockIdx.x * blockDim.x + threadIdx.x;
  if (i < n) p[i] = v;
}

__global__ void deg_accum(const int* __restrict__ dst, float* __restrict__ deg, int nE) {
  int e = blockIdx.x * blockDim.x + threadIdx.x;
  if (e < nE) atomicAdd(&deg[dst[e]], 1.0f);
}

__global__ void deg_rsqrt(float* __restrict__ deg, int n) {
  int i = blockIdx.x * blockDim.x + threadIdx.x;
  if (i < n) deg[i] = rsqrtf(deg[i]);
}

__global__ void conv_f32_to_f16(const float* __restrict__ in, _Float16* __restrict__ out, int n) {
  int i = blockIdx.x * blockDim.x + threadIdx.x;
  if (i < n) out[i] = (_Float16)in[i];
}

// W [K x N] row-major f32  ->  Wt [N x K] row-major f16 (transposed for B-frag loads)
__global__ void conv_w_t(const float* __restrict__ W, _Float16* __restrict__ Wt, int K, int N) {
  int i = blockIdx.x * blockDim.x + threadIdx.x;
  if (i < K * N) {
    int k = i / N, n = i % N;
    Wt[n * K + k] = (_Float16)W[i];
  }
}

// ---------------- WMMA GEMM: Y[n,128] = X16[n,128] * W (Wt is [128 out][128 in] f16) ----
__global__ void __launch_bounds__(256) gemm_xw_wmma(
    const _Float16* __restrict__ X16,
    const _Float16* __restrict__ Wt,
    float* __restrict__ Y,
    int nrows)
{
  __shared__ _Float16 As[16 * 128];   // one 16-row A tile shared by 8 waves
  const int m0 = blockIdx.x * 16;

  // stage A tile: 256 threads x 8 halves = 2048 halves
  {
    int t   = threadIdx.x;
    int row = t >> 4;
    int col = (t & 15) * 8;
    *(v8h*)&As[row * 128 + col] = *(const v8h*)&X16[(size_t)(m0 + row) * 128 + col];
  }
  __syncthreads();

  const int wave = threadIdx.x >> 5;   // 8 waves -> 8 column tiles of 16
  const int lane = threadIdx.x & 31;
  const int g    = lane >> 4;
  const int ln   = lane & 15;
  const int n0   = wave * 16;

  v8f c = {0.f, 0.f, 0.f, 0.f, 0.f, 0.f, 0.f, 0.f};
#pragma unroll
  for (int kk = 0; kk < 128; kk += 32) {
    // A frag: lane row = ln; halves i: K = kk + 16g + i (contiguous)
    v8h alo = *(const v8h*)&As[ln * 128 + kk + g * 16];
    v8h ahi = *(const v8h*)&As[ln * 128 + kk + g * 16 + 8];
    // B frag: lane col = n0+ln; halves i: K = kk + 16g + i (contiguous in Wt row)
    const _Float16* brow = &Wt[(size_t)(n0 + ln) * 128 + kk + g * 16];
    v8h blo = *(const v8h*)(brow);
    v8h bhi = *(const v8h*)(brow + 8);
    v16h a, b;
#pragma unroll
    for (int i = 0; i < 8; ++i) { a[i] = alo[i]; a[i + 8] = ahi[i];
                                  b[i] = blo[i]; b[i + 8] = bhi[i]; }
    c = __builtin_amdgcn_wmma_f32_16x16x32_f16(false, a, false, b, (short)0, c, false, false);
  }
  // D layout: VGPR r -> row m0 + 8g + r, col n0 + ln (coalesced across lanes)
#pragma unroll
  for (int r = 0; r < 8; ++r)
    Y[(size_t)(m0 + g * 8 + r) * 128 + n0 + ln] = c[r];
}

// ---------------- edge scatter: agg[dst] += dinv[src]*dinv[dst]*xw[src] ----------------
__global__ void __launch_bounds__(256) edge_scatter(
    const int* __restrict__ src, const int* __restrict__ dst,
    const float* __restrict__ dinv,
    const float* __restrict__ xw, float* __restrict__ agg, int nE)
{
  int gid  = blockIdx.x * blockDim.x + threadIdx.x;
  int e    = gid >> 5;            // one wave32 per edge
  int lane = gid & 31;            // lane handles 4 features
  if (e >= nE) return;
  int s = src[e], d = dst[e];
  float norm = dinv[s] * dinv[d];
  float4 v = ((const float4*)(xw + (size_t)s * 128))[lane];
  float* ad = agg + (size_t)d * 128 + lane * 4;
  atomicAdd(ad + 0, norm * v.x);
  atomicAdd(ad + 1, norm * v.y);
  atomicAdd(ad + 2, norm * v.z);
  atomicAdd(ad + 3, norm * v.w);
}

// ---------------- hidden-layer epilogue: self-loop + bias + ReLU -> f16 ----------------
__global__ void gcn_epilogue(const float* __restrict__ agg, const float* __restrict__ xw,
                             const float* __restrict__ dinv, const float* __restrict__ bias,
                             _Float16* __restrict__ h16, int n)
{
  int i = blockIdx.x * blockDim.x + threadIdx.x;
  if (i >= n * 128) return;
  int v = i >> 7, f = i & 127;
  float di  = dinv[v];
  float val = agg[i] + di * di * xw[i] + bias[f];
  val = fmaxf(val, 0.f);
  h16[i] = (_Float16)val;
}

// float atomic max with mixed signs: positives via signed-int max, negatives via unsigned min
__device__ inline void atomicMaxFloat(float* addr, float val) {
  if (val >= 0.f) atomicMax((int*)addr, __float_as_int(val));
  else            atomicMin((unsigned int*)addr, __float_as_uint(val));
}

// ---------------- final epilogue: self-loop + bias + ReLU + BN + segment max ----------
__global__ void gcn_final(const float* __restrict__ agg, const float* __restrict__ xw,
                          const float* __restrict__ dinv, const float* __restrict__ bias,
                          const float* __restrict__ gamma, const float* __restrict__ beta,
                          const float* __restrict__ rmean, const float* __restrict__ rvar,
                          const int* __restrict__ batch, float* __restrict__ gpool, int n)
{
  int i = blockIdx.x * blockDim.x + threadIdx.x;
  if (i >= n * 128) return;
  int v = i >> 7, f = i & 127;
  float di  = dinv[v];
  float val = agg[i] + di * di * xw[i] + bias[f];
  val = fmaxf(val, 0.f);
  val = (val - rmean[f]) * rsqrtf(rvar[f] + BN_EPS) * gamma[f] + beta[f];
  atomicMaxFloat(&gpool[(size_t)batch[v] * 128 + f], val);
}

// ---------------- tiny MLP head ----------------
__global__ void linear_k(const float* __restrict__ in, const float* __restrict__ W,
                         const float* __restrict__ b, float* __restrict__ out,
                         int rows, int K, int N, int doRelu)
{
  int i = blockIdx.x * blockDim.x + threadIdx.x;
  if (i >= rows * N) return;
  int r = i / N, j = i % N;
  float acc = b[j];
  for (int k = 0; k < K; ++k) acc += in[r * K + k] * W[k * N + j];
  if (doRelu) acc = fmaxf(acc, 0.f);
  out[i] = acc;
}

// ---------------- host ----------------
extern "C" void kernel_launch(void* const* d_in, const int* in_sizes, int n_in,
                              void* d_out, int out_size, void* d_ws, size_t ws_size,
                              hipStream_t stream) {
  const float* x     = (const float*)d_in[0];
  const int*   ei    = (const int*)d_in[1];
  const int*   src   = ei;
  const int*   dst   = ei + N_EDGES;
  const int*   batch = (const int*)d_in[2];
  const float* W1 = (const float*)d_in[3];  const float* b1 = (const float*)d_in[4];
  const float* W2 = (const float*)d_in[5];  const float* b2 = (const float*)d_in[6];
  const float* W3 = (const float*)d_in[7];  const float* b3 = (const float*)d_in[8];
  const float* gamma = (const float*)d_in[9];
  const float* beta  = (const float*)d_in[10];
  const float* rmean = (const float*)d_in[11];
  const float* rvar  = (const float*)d_in[12];
  const float* lw1 = (const float*)d_in[13]; const float* lb1 = (const float*)d_in[14];
  const float* lw2 = (const float*)d_in[15]; const float* lb2 = (const float*)d_in[16];
  const float* lw3 = (const float*)d_in[17]; const float* lb3 = (const float*)d_in[18];
  float* out = (float*)d_out;

  // carve workspace (256B aligned slabs)
  char* ws = (char*)d_ws;
  size_t off = 0;
  auto carve = [&](size_t bytes) -> void* {
    void* p = ws + off;
    off = (off + bytes + 255) & ~(size_t)255;
    return p;
  };
  float*     dinv  = (float*)carve((size_t)N_NODES * 4);
  _Float16*  h16   = (_Float16*)carve((size_t)N_NODES * 128 * 2);
  float*     xw    = (float*)carve((size_t)N_NODES * 128 * 4);
  float*     agg   = (float*)carve((size_t)N_NODES * 128 * 4);
  _Float16*  Wt1   = (_Float16*)carve(128 * 128 * 2);
  _Float16*  Wt2   = (_Float16*)carve(128 * 128 * 2);
  _Float16*  Wt3   = (_Float16*)carve(128 * 128 * 2);
  float*     gpool = (float*)carve((size_t)NGRAPHS * 128 * 4);
  float*     m1    = (float*)carve((size_t)NGRAPHS * 128 * 4);
  float*     m2    = (float*)carve((size_t)NGRAPHS * 64 * 4);
  (void)ws_size; (void)n_in; (void)in_sizes; (void)out_size;

  const int T = 256;
  const int NF  = N_NODES * 128;
  const int gNF = (NF + T - 1) / T;
  const int gN  = (N_NODES + T - 1) / T;
  const int gE  = (N_EDGES + T - 1) / T;
  const int gE32 = (int)(((size_t)N_EDGES * 32 + T - 1) / T);
  const int gRows = N_NODES / 16;   // 3125, exact

  // degrees: deg = 1 (self loop) + in-degree, then rsqrt
  fill_f32<<<gN, T, 0, stream>>>(dinv, 1.0f, N_NODES);
  deg_accum<<<gE, T, 0, stream>>>(dst, dinv, N_EDGES);
  deg_rsqrt<<<gN, T, 0, stream>>>(dinv, N_NODES);

  // f16 copies of features and transposed weights
  conv_f32_to_f16<<<gNF, T, 0, stream>>>(x, h16, NF);
  conv_w_t<<<(128 * 128 + T - 1) / T, T, 0, stream>>>(W1, Wt1, 128, 128);
  conv_w_t<<<(128 * 128 + T - 1) / T, T, 0, stream>>>(W2, Wt2, 128, 128);
  conv_w_t<<<(128 * 128 + T - 1) / T, T, 0, stream>>>(W3, Wt3, 128, 128);

  // ---- layer 1 ----
  gemm_xw_wmma<<<gRows, T, 0, stream>>>(h16, Wt1, xw, N_NODES);
  fill_f32<<<gNF, T, 0, stream>>>(agg, 0.f, NF);
  edge_scatter<<<gE32, T, 0, stream>>>(src, dst, dinv, xw, agg, N_EDGES);
  gcn_epilogue<<<gNF, T, 0, stream>>>(agg, xw, dinv, b1, h16, N_NODES);

  // ---- layer 2 ----
  gemm_xw_wmma<<<gRows, T, 0, stream>>>(h16, Wt2, xw, N_NODES);
  fill_f32<<<gNF, T, 0, stream>>>(agg, 0.f, NF);
  edge_scatter<<<gE32, T, 0, stream>>>(src, dst, dinv, xw, agg, N_EDGES);
  gcn_epilogue<<<gNF, T, 0, stream>>>(agg, xw, dinv, b2, h16, N_NODES);

  // ---- layer 3 + BN + segment max ----
  gemm_xw_wmma<<<gRows, T, 0, stream>>>(h16, Wt3, xw, N_NODES);
  fill_f32<<<gNF, T, 0, stream>>>(agg, 0.f, NF);
  edge_scatter<<<gE32, T, 0, stream>>>(src, dst, dinv, xw, agg, N_EDGES);
  fill_f32<<<(NGRAPHS * 128 + T - 1) / T, T, 0, stream>>>(gpool, -1e30f, NGRAPHS * 128);
  gcn_final<<<gNF, T, 0, stream>>>(agg, xw, dinv, b3, gamma, beta, rmean, rvar,
                                   batch, gpool, N_NODES);

  // ---- MLP head (tiny) ----
  linear_k<<<(NGRAPHS * 128 + T - 1) / T, T, 0, stream>>>(gpool, lw1, lb1, m1,
                                                          NGRAPHS, 128, 128, 1);
  linear_k<<<(NGRAPHS * 64 + T - 1) / T, T, 0, stream>>>(m1, lw2, lb2, m2,
                                                         NGRAPHS, 128, 64, 1);
  linear_k<<<(NGRAPHS * NCLS + T - 1) / T, T, 0, stream>>>(m2, lw3, lb3, out,
                                                           NGRAPHS, 64, NCLS, 0);
}
